// Dynamic_Influence_Model_86449101734283
// MI455X (gfx1250) — matrix operand order repaired and checked by reading, not compiled, not run on hardware
//
#include <hip/hip_runtime.h>
#include <hip/hip_bf16.h>
#include <cstdint>

#define TT 16
#define NNODES_ 50000
#define DD 128
#define BB 64
#define RR 3
#define NBB 64
#define HH 128
#define G4 512        // 4*H
#define MM 4096       // B*NB
#define OUTD 64
#define INFD 64
#define H2 256        // 2*H

typedef __attribute__((ext_vector_type(16))) __bf16 v16bf;
typedef __attribute__((ext_vector_type(8)))  __bf16 v8bf;
typedef __attribute__((ext_vector_type(8)))  float  v8f;

// ---- WMMA fragment helpers (16x16x32 bf16, contiguous k = 16h+s in both A and B) ----

// A matrix 16x32 (MxK), row-major source with leading dim `ld`.
// lane<16: elems 0..15 = K0..15 ; lane>=16: K16..31
__device__ __forceinline__ v16bf load_frag_a(const __bf16* base, int ld, int row,
                                             int k0, int lane) {
  const int koff = (lane & 16) ? 16 : 0;
  const __bf16* p = base + (size_t)row * ld + k0 + koff;
  v8bf lo = *(const v8bf*)(p);
  v8bf hi = *(const v8bf*)(p + 8);
  v16bf f;
#pragma unroll
  for (int i = 0; i < 8; ++i) { f[i] = lo[i]; f[i + 8] = hi[i]; }
  return f;
}

// B matrix 32x16 (KxN) with B[k][n] = W[n][k], W row-major, leading dim `ld`.
// lane n holds 16 contiguous K values starting at (lane>=16 ? 16 : 0).
__device__ __forceinline__ v16bf load_frag_b(const __bf16* Wrow, int ld, int row,
                                             int k0, int lane) {
  const int koff = (lane & 16) ? 16 : 0;
  return *(const v16bf*)(Wrow + (size_t)row * ld + k0 + koff);
}

__device__ __forceinline__ v8f wmma_bf16(v16bf a, v16bf b, v8f c) {
  return __builtin_amdgcn_wmma_f32_16x16x32_bf16(
      /*neg_a=*/false, a, /*neg_b=*/false, b,
      /*c_mod=*/(short)0, c, /*reuse_a=*/false, /*reuse_b=*/false);
}

__device__ __forceinline__ float sigm(float x) { return 1.0f / (1.0f + __expf(-x)); }

// ---- utility kernels ----

__global__ void zero_kernel(float* p, int n) {
  int i = blockIdx.x * blockDim.x + threadIdx.x;
  if (i < n) p[i] = 0.0f;
}

__global__ void f2bf_kernel(const float* __restrict__ src, __bf16* __restrict__ dst, int n) {
  int i = blockIdx.x * blockDim.x + threadIdx.x;
  if (i < n) dst[i] = (__bf16)src[i];
}

// WcT[j*OUT + k] = ((W1+W2) @ Wrel)[k][j]   (transposed so it loads as a WMMA B-matrix)
__global__ void wc_kernel(const float* __restrict__ W1, const float* __restrict__ W2,
                          const float* __restrict__ Wrel, __bf16* __restrict__ wcT) {
  int idx = blockIdx.x * blockDim.x + threadIdx.x;
  if (idx >= OUTD * INFD) return;
  int k = idx >> 6;   // OUT index
  int j = idx & 63;   // INF index
  float s = 0.0f;
  for (int q = 0; q < INFD; ++q)
    s += (W1[k * INFD + q] + W2[k * INFD + q]) * Wrel[q * INFD + j];
  wcT[j * OUTD + k] = (__bf16)s;
}

// one wave per (t, seq): gather + L2-normalize + bf16 convert
__global__ void gather_kernel(const float* __restrict__ emb, const int* __restrict__ align,
                              const int* __restrict__ neigh, int r, __bf16* __restrict__ xbf) {
  const int gw = blockIdx.x * (blockDim.x >> 5) + (threadIdx.x >> 5);
  const int lane = threadIdx.x & 31;
  if (gw >= TT * MM) return;
  const int t = gw >> 12;        // / MM
  const int s = gw & (MM - 1);
  const int b = s >> 6;
  const int nb = s & 63;
  const int a = neigh[(b * RR + r) * NBB + nb];
  const int node = align[a * TT + t];
  const float* e = emb + ((size_t)t * NNODES_ + node) * DD;
  float v[4]; float ss = 0.0f;
#pragma unroll
  for (int i = 0; i < 4; ++i) { v[i] = e[lane + 32 * i]; ss += v[i] * v[i]; }
#pragma unroll
  for (int mk = 16; mk >= 1; mk >>= 1) ss += __shfl_xor(ss, mk, 32);
  const float scale = 1.0f / fmaxf(sqrtf(ss), 1e-12f);
  __bf16* o = xbf + ((size_t)t * MM + s) * DD;
#pragma unroll
  for (int i = 0; i < 4; ++i) o[lane + 32 * i] = (__bf16)(v[i] * scale);
}

// ---- fused bidirectional LSTM: one WG = 16 rows of M, all T steps ----
__launch_bounds__(256)
__global__ void lstm_kernel(const __bf16* __restrict__ xbf,
                            const __bf16* __restrict__ wihF, const __bf16* __restrict__ whhF,
                            const __bf16* __restrict__ wihB, const __bf16* __restrict__ whhB,
                            const float* __restrict__ biF, const float* __restrict__ bhF,
                            const float* __restrict__ biB, const float* __restrict__ bhB,
                            __bf16* __restrict__ houtF, __bf16* __restrict__ houtB) {
  const int dir  = blockIdx.y;
  const int m0   = blockIdx.x * 16;
  const int tid  = threadIdx.x;
  const int wv   = tid >> 5;
  const int lane = tid & 31;
  const int l15  = lane & 15;

  const __bf16* Wih = dir ? wihB : wihF;
  const __bf16* Whh = dir ? whhB : whhF;
  const float*  bi  = dir ? biB : biF;
  const float*  bh  = dir ? bhB : bhF;
  __bf16* hout      = dir ? houtB : houtF;

  __shared__ float  sZ[16 * G4];   // 32 KB gate pre-activations
  __shared__ __bf16 sH[16 * HH];   // 4 KB recurrent state (bf16)

  for (int i = tid; i < 16 * HH; i += 256) sH[i] = (__bf16)0.0f;
  __syncthreads();

  float bv[4];
#pragma unroll
  for (int i = 0; i < 4; ++i) {
    int n = (wv * 4 + i) * 16 + l15;
    bv[i] = bi[n] + bh[n];
  }

  const int gm  = tid >> 4;         // gate-phase row 0..15
  const int gj0 = (tid & 15) * 8;   // gate-phase col block
  float cst[8];
#pragma unroll
  for (int jj = 0; jj < 8; ++jj) cst[jj] = 0.0f;

  for (int t = 0; t < TT; ++t) {
    const int xt = dir ? (TT - 1 - t) : t;
    const __bf16* xrow = xbf + ((size_t)xt * MM + m0) * DD;

    v8f acc[4];
#pragma unroll
    for (int i = 0; i < 4; ++i)
#pragma unroll
      for (int e = 0; e < 8; ++e) acc[i][e] = bv[i];

    // z += x_t @ Wih^T   (K = D = 128)
#pragma unroll
    for (int kb = 0; kb < 4; ++kb) {
      const int k0 = kb * 32;
      v16bf a = load_frag_a(xrow, DD, l15, k0, lane);
#pragma unroll
      for (int i = 0; i < 4; ++i) {
        const int n0 = (wv * 4 + i) * 16;
        v16bf bfr = load_frag_b(Wih, DD, n0 + l15, k0, lane);
        acc[i] = wmma_bf16(a, bfr, acc[i]);
      }
    }
    // z += h @ Whh^T   (K = H = 128, h from LDS)
#pragma unroll
    for (int kb = 0; kb < 4; ++kb) {
      const int k0 = kb * 32;
      v16bf a = load_frag_a(sH, HH, l15, k0, lane);
#pragma unroll
      for (int i = 0; i < 4; ++i) {
        const int n0 = (wv * 4 + i) * 16;
        v16bf bfr = load_frag_b(Whh, HH, n0 + l15, k0, lane);
        acc[i] = wmma_bf16(a, bfr, acc[i]);
      }
    }

    const int rbase = (lane & 16) ? 8 : 0;
#pragma unroll
    for (int i = 0; i < 4; ++i) {
      const int n0 = (wv * 4 + i) * 16;
#pragma unroll
      for (int e = 0; e < 8; ++e)
        sZ[(rbase + e) * G4 + n0 + l15] = acc[i][e];
    }
    __syncthreads();

    // gates + recurrence (c in registers, h back into LDS as bf16)
#pragma unroll
    for (int jj = 0; jj < 8; ++jj) {
      const int j = gj0 + jj;
      const float iv = sZ[gm * G4 + j];
      const float fv = sZ[gm * G4 + HH + j];
      const float gv = sZ[gm * G4 + 2 * HH + j];
      const float ov = sZ[gm * G4 + 3 * HH + j];
      const float c = sigm(fv) * cst[jj] + sigm(iv) * tanhf(gv);
      const float h = sigm(ov) * tanhf(c);
      cst[jj] = c;
      sH[gm * HH + j] = (__bf16)h;
      if (t == TT - 1)
        hout[(size_t)(m0 + gm) * HH + j] = (__bf16)fmaxf(h, 0.0f);  // relu fused
    }
    __syncthreads();
  }
}

// ---- fc + influence + per-b reduction: WG = 128 rows, wave = 16x64 tile ----
__launch_bounds__(256)
__global__ void fc_inf_kernel(const __bf16* __restrict__ hb, const __bf16* __restrict__ hf,
                              const __bf16* __restrict__ fcw, const float* __restrict__ fcb,
                              const __bf16* __restrict__ wcT, float* __restrict__ total) {
  const int tid = threadIdx.x;
  const int wv = tid >> 5;
  const int lane = tid & 31;
  const int l15 = lane & 15;
  const int gm0 = blockIdx.x * 128 + wv * 16;

  __shared__ __bf16 sO[8 * 16 * OUTD];   // 16 KB: per-wave o tiles in bf16
  __bf16* so = sO + wv * (16 * OUTD);

  v8f acc[4];
#pragma unroll
  for (int i = 0; i < 4; ++i) {
    const float b = fcb[i * 16 + l15];
#pragma unroll
    for (int e = 0; e < 8; ++e) acc[i][e] = b;
  }

  // o = relu(hs) @ fc_W^T + fc_b ; hs = [h_b | h_f], K = 256
#pragma unroll
  for (int kb = 0; kb < 8; ++kb) {
    const int k0 = kb * 32;
    const __bf16* abase = (k0 < HH) ? hb : hf;
    v16bf a = load_frag_a(abase, HH, gm0 + l15, k0 & (HH - 1), lane);
#pragma unroll
    for (int i = 0; i < 4; ++i) {
      v16bf bfr = load_frag_b(fcw, H2, i * 16 + l15, k0, lane);
      acc[i] = wmma_bf16(a, bfr, acc[i]);
    }
  }

  const int rbase = (lane & 16) ? 8 : 0;
#pragma unroll
  for (int i = 0; i < 4; ++i)
#pragma unroll
    for (int e = 0; e < 8; ++e)
      so[(rbase + e) * OUTD + i * 16 + l15] = (__bf16)acc[i][e];
  __syncthreads();

  // inf = o @ Wc   (K = 64)
  v8f ai[4];
#pragma unroll
  for (int i = 0; i < 4; ++i)
#pragma unroll
    for (int e = 0; e < 8; ++e) ai[i][e] = 0.0f;

#pragma unroll
  for (int kb = 0; kb < 2; ++kb) {
    const int k0 = kb * 32;
    v16bf a = load_frag_a(so, OUTD, l15, k0, lane);
#pragma unroll
    for (int i = 0; i < 4; ++i) {
      v16bf bfr = load_frag_b(wcT, OUTD, i * 16 + l15, k0, lane);
      ai[i] = wmma_bf16(a, bfr, ai[i]);
    }
  }

  // all 16 rows of this wave share one b (NB=64): reduce rows then atomicAdd
  const int b = gm0 >> 6;
#pragma unroll
  for (int i = 0; i < 4; ++i) {
    float s = 0.0f;
#pragma unroll
    for (int e = 0; e < 8; ++e) s += ai[i][e];
    s += __shfl_xor(s, 16, 32);
    if (lane < 16) atomicAdd(&total[b * INFD + i * 16 + l15], s);
  }
}

extern "C" void kernel_launch(void* const* d_in, const int* in_sizes, int n_in,
                              void* d_out, int out_size, void* d_ws, size_t ws_size,
                              hipStream_t stream) {
  (void)in_sizes; (void)n_in; (void)out_size; (void)ws_size;
  const float* emb   = (const float*)d_in[0];
  const int*   align = (const int*)d_in[1];
  const int*   neigh = (const int*)d_in[2];
  const float* WihF  = (const float*)d_in[3];
  const float* WhhF  = (const float*)d_in[4];
  const float* bihF  = (const float*)d_in[5];
  const float* bhhF  = (const float*)d_in[6];
  const float* WihB  = (const float*)d_in[7];
  const float* WhhB  = (const float*)d_in[8];
  const float* bihB  = (const float*)d_in[9];
  const float* bhhB  = (const float*)d_in[10];
  const float* fcW   = (const float*)d_in[11];
  const float* fcb   = (const float*)d_in[12];
  const float* W1    = (const float*)d_in[13];
  const float* W2    = (const float*)d_in[14];
  const float* Wrel  = (const float*)d_in[15];
  float* total = (float*)d_out;

  uint8_t* ws = (uint8_t*)d_ws;
  size_t off = 0;
  auto alloc = [&](size_t bytes) -> void* {
    void* p = ws + off;
    off = (off + bytes + 255) & ~(size_t)255;
    return p;
  };
  __bf16* xbf    = (__bf16*)alloc((size_t)TT * MM * DD * 2);   // 16 MB
  __bf16* wihf   = (__bf16*)alloc((size_t)G4 * DD * 2);
  __bf16* whhf   = (__bf16*)alloc((size_t)G4 * HH * 2);
  __bf16* wihb   = (__bf16*)alloc((size_t)G4 * DD * 2);
  __bf16* whhb   = (__bf16*)alloc((size_t)G4 * HH * 2);
  __bf16* fcwb   = (__bf16*)alloc((size_t)OUTD * H2 * 2);
  __bf16* wcT    = (__bf16*)alloc((size_t)INFD * OUTD * 2);
  __bf16* hbuf_f = (__bf16*)alloc((size_t)MM * HH * 2);
  __bf16* hbuf_b = (__bf16*)alloc((size_t)MM * HH * 2);

  zero_kernel<<<(BB * INFD + 255) / 256, 256, 0, stream>>>(total, BB * INFD);

  for (int r = 0; r < RR; ++r) {
    const int WN = G4 * DD;
    f2bf_kernel<<<(WN + 255) / 256, 256, 0, stream>>>(WihF + (size_t)r * WN, wihf, WN);
    f2bf_kernel<<<(G4 * HH + 255) / 256, 256, 0, stream>>>(WhhF + (size_t)r * G4 * HH, whhf, G4 * HH);
    f2bf_kernel<<<(WN + 255) / 256, 256, 0, stream>>>(WihB + (size_t)r * WN, wihb, WN);
    f2bf_kernel<<<(G4 * HH + 255) / 256, 256, 0, stream>>>(WhhB + (size_t)r * G4 * HH, whhb, G4 * HH);
    f2bf_kernel<<<(OUTD * H2 + 255) / 256, 256, 0, stream>>>(fcW + (size_t)r * OUTD * H2, fcwb, OUTD * H2);
    wc_kernel<<<(OUTD * INFD + 255) / 256, 256, 0, stream>>>(
        W1 + (size_t)r * OUTD * INFD, W2 + (size_t)r * OUTD * INFD,
        Wrel + (size_t)r * INFD * INFD, wcT);

    gather_kernel<<<(TT * MM) / 8, 256, 0, stream>>>(emb, align, neigh, r, xbf);

    dim3 lg(MM / 16, 2);
    lstm_kernel<<<lg, 256, 0, stream>>>(xbf, wihf, whhf, wihb, whhb,
                                        bihF + r * G4, bhhF + r * G4,
                                        bihB + r * G4, bhhB + r * G4,
                                        hbuf_f, hbuf_b);

    fc_inf_kernel<<<MM / 128, 256, 0, stream>>>(hbuf_b, hbuf_f, fcwb,
                                                fcb + r * OUTD, wcT, total);
  }
}
